// Glm4MoeAttention_25245817766050
// MI455X (gfx1250) — compile-verified, matrix operand order checked by reading, not run on hardware
//
#include <hip/hip_runtime.h>
#include <hip/hip_bf16.h>

#define SDIM 2048
#define HIDD 4096
#define NH   32
#define NKV  8
#define DH   128
#define ROTD 64

typedef __attribute__((ext_vector_type(16))) __bf16 v16bf;
typedef __attribute__((ext_vector_type(8)))  __bf16 v8bf;
typedef __attribute__((ext_vector_type(4)))  __bf16 v4bf;
typedef __attribute__((ext_vector_type(8)))  float  v8f;
typedef __attribute__((ext_vector_type(4)))  float  v4f;

union FragU { v16bf v; v8bf h[2]; };

__device__ __forceinline__ v16bf make_frag(v8bf a, v8bf b) {
  FragU u; u.h[0] = a; u.h[1] = b; return u.v;
}

// LDS byte offset of a generic pointer into a __shared__ object
__device__ __forceinline__ uint32_t lds_addr_of(const void* p) {
  return (uint32_t)(uintptr_t)(__attribute__((address_space(3))) const void*)p;
}

// CDNA5 async copy: 16 bytes/lane global -> LDS, tracked by ASYNCcnt
__device__ __forceinline__ void async_copy16(uint32_t lds_off, const void* gptr) {
  asm volatile("global_load_async_to_lds_b128 %0, %1, off"
               :: "v"(lds_off), "v"((unsigned long long)(uintptr_t)gptr)
               : "memory");
}

template <int N>
__device__ __forceinline__ void wait_async() {
  asm volatile("s_wait_asynccnt %0" :: "n"(N) : "memory");
}

// ---------------------------------------------------------------- cvt f32->bf16
__global__ void cvt_f32_bf16(const float* __restrict__ src,
                             __bf16* __restrict__ dst, int n4) {
  int i = blockIdx.x * blockDim.x + threadIdx.x;
  int stride = gridDim.x * blockDim.x;
  for (; i < n4; i += stride) {
    v4f x = ((const v4f*)src)[i];
    v4bf y;
    y.x = (__bf16)x.x; y.y = (__bf16)x.y; y.z = (__bf16)x.z; y.w = (__bf16)x.w;
    ((v4bf*)dst)[i] = y;
  }
}

// ---------------------------------------------------------------- bf16 WMMA GEMM
// C[M,N] (f32, row-major) = A[M,K] (bf16 row-major) * B[N,K]^T (bf16 row-major)
// block = 256 threads = 8 waves; tile 128x128x32; wave tile 64x32 (4x2 WMMA)
// Double-buffered GLOBAL_LOAD_ASYNC_TO_LDS_B128 staging (s_wait_asynccnt 4
// overlaps next-tile DMA with current-tile WMMA) + global_prefetch.
__global__ void __launch_bounds__(256)
gemm_bf16_nt(const __bf16* __restrict__ A, const __bf16* __restrict__ B,
             float* __restrict__ C, int M, int N, int K) {
  __shared__ alignas(16) __bf16 As[2][128 * 32];
  __shared__ alignas(16) __bf16 Bs[2][128 * 32];
  const int tid  = threadIdx.x;
  const int lane = tid & 31;
  const int w    = tid >> 5;
  const int ln   = lane & 15;
  const int hi   = lane >> 4;
  const int wm   = w >> 2;      // 0..1
  const int wn   = w & 3;       // 0..3
  const int m0   = blockIdx.y * 128;
  const int n0   = blockIdx.x * 128;
  const int lrow  = tid >> 1;            // 0..127
  const int lhalf = (tid & 1) * 16;      // 0 or 16 elements

  v8f acc[4][2];
  #pragma unroll
  for (int mt = 0; mt < 4; ++mt)
    #pragma unroll
    for (int nt = 0; nt < 2; ++nt)
      #pragma unroll
      for (int j = 0; j < 8; ++j) acc[mt][nt][j] = 0.0f;

  const __bf16* Ag = A + (size_t)(m0 + lrow) * K + lhalf;
  const __bf16* Bg = B + (size_t)(n0 + lrow) * K + lhalf;
  const uint32_t lA = lds_addr_of(&As[0][0] + lrow * 32 + lhalf);
  const uint32_t lB = lds_addr_of(&Bs[0][0] + lrow * 32 + lhalf);
  const uint32_t BUFB = 128 * 32 * 2;    // bytes per LDS buffer

  // prologue: stage tile 0 into buffer 0
  async_copy16(lA,      Ag);
  async_copy16(lA + 16, Ag + 8);
  async_copy16(lB,      Bg);
  async_copy16(lB + 16, Bg + 8);

  for (int k0 = 0; k0 < K; k0 += 32) {
    const int p = (k0 >> 5) & 1;
    if (k0 + 32 < K) {
      // stage tile k+1 into the other buffer while tile k computes
      const uint32_t oA = lA + (p ^ 1) * BUFB;
      const uint32_t oB = lB + (p ^ 1) * BUFB;
      async_copy16(oA,      Ag + k0 + 32);
      async_copy16(oA + 16, Ag + k0 + 40);
      async_copy16(oB,      Bg + k0 + 32);
      async_copy16(oB + 16, Bg + k0 + 40);
      __builtin_prefetch(Ag + k0 + 64, 0, 0);
      __builtin_prefetch(Bg + k0 + 64, 0, 0);
      wait_async<4>();   // older group of 4 has landed (in-order completion)
    } else {
      wait_async<0>();
    }
    __syncthreads();

    const __bf16* Asp = &As[p][0];
    const __bf16* Bsp = &Bs[p][0];
    v16bf af[4], bfr[2];
    #pragma unroll
    for (int mt = 0; mt < 4; ++mt) {
      const __bf16* pa = Asp + (wm * 64 + mt * 16 + ln) * 32 + 8 * hi;
      af[mt] = make_frag(*(const v8bf*)pa, *(const v8bf*)(pa + 16));
    }
    #pragma unroll
    for (int nt = 0; nt < 2; ++nt) {
      const __bf16* pb = Bsp + (wn * 32 + nt * 16 + ln) * 32 + 8 * hi;
      bfr[nt] = make_frag(*(const v8bf*)pb, *(const v8bf*)(pb + 16));
    }
    #pragma unroll
    for (int mt = 0; mt < 4; ++mt)
      #pragma unroll
      for (int nt = 0; nt < 2; ++nt)
        acc[mt][nt] = __builtin_amdgcn_wmma_f32_16x16x32_bf16(
            false, af[mt], false, bfr[nt], (short)0, acc[mt][nt], false, false);
    __syncthreads();
  }

  #pragma unroll
  for (int mt = 0; mt < 4; ++mt)
    #pragma unroll
    for (int nt = 0; nt < 2; ++nt)
      #pragma unroll
      for (int j = 0; j < 8; ++j)
        C[(size_t)(m0 + wm * 64 + mt * 16 + 8 * hi + j) * N +
          (n0 + wn * 32 + nt * 16 + ln)] = acc[mt][nt][j];
}

// ---------------------------------------------------------------- RMSNorm + RoPE
// one wave per (slot, s): slots 0..31 = Q heads, 32..39 = K heads, 40..47 = V heads
__global__ void __launch_bounds__(128)
normrope_kernel(const float* __restrict__ qkv, const float* __restrict__ cosb,
                const float* __restrict__ sinb, const float* __restrict__ qw,
                const float* __restrict__ kw, __bf16* __restrict__ Qh,
                __bf16* __restrict__ Kh, __bf16* __restrict__ Vt) {
  const int task = blockIdx.x * 4 + (threadIdx.x >> 5);
  const int lane = threadIdx.x & 31;
  const int slot = task >> 11;     // 0..47
  const int s    = task & 2047;

  const float* row;
  int h;
  if (slot < NH)            { h = slot;            row = qkv + (size_t)s * 6144 + h * DH; }
  else if (slot < NH + NKV) { h = slot - NH;       row = qkv + (size_t)s * 6144 + 4096 + h * DH; }
  else                      { h = slot - NH - NKV; row = qkv + (size_t)s * 6144 + 5120 + h * DH; }

  v4f x = *(const v4f*)(row + lane * 4);

  if (slot < NH + NKV) {
    float ss = x.x * x.x + x.y * x.y + x.z * x.z + x.w * x.w;
    #pragma unroll
    for (int off = 16; off > 0; off >>= 1) ss += __shfl_xor(ss, off, 32);
    float r = rsqrtf(ss * (1.0f / 128.0f) + 1e-6f);
    const float* wgt = (slot < NH) ? qw : kw;
    v4f wv = *(const v4f*)(wgt + lane * 4);
    float y[4] = { x.x * r * wv.x, x.y * r * wv.y, x.z * r * wv.z, x.w * r * wv.w };
    float oy[4];
    #pragma unroll
    for (int i = 0; i < 4; ++i) oy[i] = __shfl_xor(y[i], 8, 32);
    if (lane < 16) {  // d in [0,64): rotary
      const float* cp = cosb + (size_t)s * ROTD + lane * 4;
      const float* sp = sinb + (size_t)s * ROTD + lane * 4;
      #pragma unroll
      for (int i = 0; i < 4; ++i) {
        float c = cp[i], sn = sp[i];
        y[i] = (lane < 8) ? (y[i] * c - oy[i] * sn)   // x1*cos - x2*sin
                          : (y[i] * c + oy[i] * sn);  // x2*cos + x1*sin
      }
    }
    __bf16* dst = (slot < NH) ? (Qh + ((size_t)h * SDIM + s) * DH + lane * 4)
                              : (Kh + ((size_t)h * SDIM + s) * DH + lane * 4);
    v4bf o;
    o.x = (__bf16)y[0]; o.y = (__bf16)y[1]; o.z = (__bf16)y[2]; o.w = (__bf16)y[3];
    *(v4bf*)dst = o;
  } else {
    // V: plain convert, store transposed [h][d][s] for contiguous PV B-fragments
    float y[4] = { x.x, x.y, x.z, x.w };
    #pragma unroll
    for (int i = 0; i < 4; ++i)
      Vt[((size_t)h * DH + lane * 4 + i) * SDIM + s] = (__bf16)y[i];
  }
}

// ---------------------------------------------------------------- flash attention
// one wave per 16-row q-tile; 4 waves / block; causal; online softmax
__global__ void __launch_bounds__(128)
attn_kernel(const __bf16* __restrict__ Qh, const __bf16* __restrict__ Kh,
            const __bf16* __restrict__ Vt, __bf16* __restrict__ Ao) {
  __shared__ alignas(16) __bf16 Plds[4][16 * 32];
  const int w    = threadIdx.x >> 5;
  const int lane = threadIdx.x & 31;
  const int ln   = lane & 15;
  const int hi   = lane >> 4;
  const int h    = blockIdx.y;
  const int hkv  = h >> 2;                         // rep = H/HKV = 4
  const int qm   = (blockIdx.x * 4 + w) * 16;
  const float scale = 0.08838834764831845f;        // 1/sqrt(128)

  // Q fragments: 4 A-frags covering D=128
  const __bf16* Qrow = Qh + ((size_t)h * SDIM + qm + ln) * DH;
  v16bf qf[4];
  #pragma unroll
  for (int kt = 0; kt < 4; ++kt) {
    const __bf16* p = Qrow + kt * 32 + 8 * hi;
    qf[kt] = make_frag(*(const v8bf*)p, *(const v8bf*)(p + 16));
  }

  v8f o[8];
  float m[8], l[8];
  #pragma unroll
  for (int dt = 0; dt < 8; ++dt)
    #pragma unroll
    for (int j = 0; j < 8; ++j) o[dt][j] = 0.0f;
  #pragma unroll
  for (int j = 0; j < 8; ++j) { m[j] = -3.0e38f; l[j] = 0.0f; }

  const int nkb = (qm + 47) >> 5;                  // causal key-block count
  for (int kb = 0; kb < nkb; ++kb) {
    // ---- S = Q K^T (16 x 32), two 16x16 accumulators
    v8f s0, s1;
    #pragma unroll
    for (int j = 0; j < 8; ++j) { s0[j] = 0.0f; s1[j] = 0.0f; }
    const __bf16* Kb0 = Kh + ((size_t)hkv * SDIM + kb * 32 + ln) * DH;
    const __bf16* Kb1 = Kb0 + 16 * DH;
    #pragma unroll
    for (int kt = 0; kt < 4; ++kt) {
      const __bf16* p0 = Kb0 + kt * 32 + 8 * hi;
      const __bf16* p1 = Kb1 + kt * 32 + 8 * hi;
      v16bf k0 = make_frag(*(const v8bf*)p0, *(const v8bf*)(p0 + 16));
      v16bf k1 = make_frag(*(const v8bf*)p1, *(const v8bf*)(p1 + 16));
      s0 = __builtin_amdgcn_wmma_f32_16x16x32_bf16(false, qf[kt], false, k0,
                                                   (short)0, s0, false, false);
      s1 = __builtin_amdgcn_wmma_f32_16x16x32_bf16(false, qf[kt], false, k1,
                                                   (short)0, s1, false, false);
    }
    // ---- online softmax per row (row = qm + 8*hi + j, cols in 16-lane group)
    #pragma unroll
    for (int j = 0; j < 8; ++j) {
      const int rrow = qm + 8 * hi + j;
      const int c0 = kb * 32 + ln, c1 = c0 + 16;
      float f0 = (c0 <= rrow) ? s0[j] * scale : -3.0e38f;
      float f1 = (c1 <= rrow) ? s1[j] * scale : -3.0e38f;
      float t = fmaxf(f0, f1);
      #pragma unroll
      for (int off = 8; off > 0; off >>= 1) t = fmaxf(t, __shfl_xor(t, off, 32));
      const float mn   = fmaxf(m[j], t);
      const float corr = __expf(m[j] - mn);
      const float p0   = __expf(f0 - mn);
      const float p1   = __expf(f1 - mn);
      float rs = p0 + p1;
      #pragma unroll
      for (int off = 8; off > 0; off >>= 1) rs += __shfl_xor(rs, off, 32);
      l[j] = l[j] * corr + rs;
      m[j] = mn;
      #pragma unroll
      for (int dt = 0; dt < 8; ++dt) o[dt][j] *= corr;
      Plds[w][(8 * hi + j) * 32 + ln]      = (__bf16)p0;
      Plds[w][(8 * hi + j) * 32 + 16 + ln] = (__bf16)p1;
    }
    asm volatile("s_wait_dscnt 0" ::: "memory");   // P C-layout -> A-frag re-read
    const __bf16* pp = &Plds[w][ln * 32 + 8 * hi];
    v16bf pf = make_frag(*(const v8bf*)pp, *(const v8bf*)(pp + 16));
    // ---- O += P V  (V transposed: contiguous k per fixed d)
    #pragma unroll
    for (int dt = 0; dt < 8; ++dt) {
      const __bf16* vp = Vt + ((size_t)hkv * DH + dt * 16 + ln) * SDIM + kb * 32 + 8 * hi;
      v16bf vf = make_frag(*(const v8bf*)vp, *(const v8bf*)(vp + 16));
      o[dt] = __builtin_amdgcn_wmma_f32_16x16x32_bf16(false, pf, false, vf,
                                                      (short)0, o[dt], false, false);
    }
  }

  // ---- normalize and write attn output [s, h*128 + d] bf16
  #pragma unroll
  for (int dt = 0; dt < 8; ++dt)
    #pragma unroll
    for (int j = 0; j < 8; ++j) {
      float val = o[dt][j] / l[j];
      Ao[(size_t)(qm + 8 * hi + j) * (NH * DH) + h * DH + dt * 16 + ln] = (__bf16)val;
    }
}

// ---------------------------------------------------------------- launch
extern "C" void kernel_launch(void* const* d_in, const int* in_sizes, int n_in,
                              void* d_out, int out_size, void* d_ws, size_t ws_size,
                              hipStream_t stream) {
  const float* hidden = (const float*)d_in[0];
  const float* cosb   = (const float*)d_in[1];
  const float* sinb   = (const float*)d_in[2];
  const float* Wq     = (const float*)d_in[3];
  const float* Wk     = (const float*)d_in[4];
  const float* Wv     = (const float*)d_in[5];
  const float* Wo     = (const float*)d_in[6];
  const float* qw     = (const float*)d_in[7];
  const float* kw     = (const float*)d_in[8];

  char* ws = (char*)d_ws;
  size_t off = 0;
  auto take = [&](size_t bytes) -> char* {
    char* p = ws + off;
    off += (bytes + 255) & ~(size_t)255;
    return p;
  };
  __bf16* hid_bf  = (__bf16*)take((size_t)SDIM * HIDD * 2);
  __bf16* wqkv_bf = (__bf16*)take((size_t)6144 * HIDD * 2);
  __bf16* wo_bf   = (__bf16*)take((size_t)HIDD * HIDD * 2);
  float*  qkv_f   = (float*) take((size_t)SDIM * 6144 * 4);
  __bf16* qh      = (__bf16*)take((size_t)NH  * SDIM * DH * 2);
  __bf16* kh      = (__bf16*)take((size_t)NKV * SDIM * DH * 2);
  __bf16* vt      = (__bf16*)take((size_t)NKV * SDIM * DH * 2);
  __bf16* ao      = (__bf16*)take((size_t)SDIM * NH * DH * 2);

  // fp32 -> bf16 (weights concatenated row-wise into one [6144, 4096] matrix)
  cvt_f32_bf16<<<2048, 256, 0, stream>>>(hidden, hid_bf, SDIM * HIDD / 4);
  cvt_f32_bf16<<<2048, 256, 0, stream>>>(Wq, wqkv_bf, 4096 * 4096 / 4);
  cvt_f32_bf16<<<1024, 256, 0, stream>>>(Wk, wqkv_bf + (size_t)4096 * 4096, 1024 * 4096 / 4);
  cvt_f32_bf16<<<1024, 256, 0, stream>>>(Wv, wqkv_bf + (size_t)5120 * 4096, 1024 * 4096 / 4);
  cvt_f32_bf16<<<2048, 256, 0, stream>>>(Wo, wo_bf, 4096 * 4096 / 4);

  // fused QKV projection: [2048, 6144] fp32
  gemm_bf16_nt<<<dim3(48, 16), 256, 0, stream>>>(hid_bf, wqkv_bf, qkv_f,
                                                 SDIM, 6144, HIDD);
  // RMSNorm + RoPE + layout (V transposed)
  normrope_kernel<<<(48 * SDIM) / 4, 128, 0, stream>>>(qkv_f, cosb, sinb, qw, kw,
                                                       qh, kh, vt);
  // causal flash attention
  attn_kernel<<<dim3(SDIM / 64, NH), 128, 0, stream>>>(qh, kh, vt, ao);
  // output projection -> fp32 d_out
  gemm_bf16_nt<<<dim3(32, 16), 256, 0, stream>>>(ao, wo_bf, (float*)d_out,
                                                 SDIM, HIDD, HIDD);
}
